// GNNLinkPredictor_22222160789513
// MI455X (gfx1250) — compile-verified, matrix operand order checked by reading, not run on hardware
//
#include <hip/hip_runtime.h>
#include <hip/hip_bf16.h>

typedef __attribute__((ext_vector_type(16))) _Float16 v16h;
typedef __attribute__((ext_vector_type(8)))  _Float16 v8h;
typedef __attribute__((ext_vector_type(4)))  _Float16 v4h;
typedef __attribute__((ext_vector_type(8)))  float    v8f;

#define NNODES 100000
#define NEDGES 640000
#define INC    128
#define HIDC   128
#define OUTC   64
#define KTOT   256           // [mean | feat] concatenated K
#define A_STRIDE 264         // 256 + 8 halves pad; row stride = 528 B = 33*16 (keeps b128 aligned)

// ---------------- workspace layout (bytes) ----------------
// agg  : f32 [NNODES][128]   scatter accumulator (L2-resident, 51.2 MB)
// cnt  : f32 [NNODES]        in-degree counts
// h    : f16 [NNODES][128]   hidden activations
// wc1  : f16 [128][256]      [W1_l | W1_r] packed K-major per output row
// wc2  : f16 [64][256]       [W2_l | W2_r]
static constexpr size_t AGG_OFF = 0;
static constexpr size_t CNT_OFF = (size_t)NNODES * 128 * 4;          // 51,200,000
static constexpr size_t H_OFF   = CNT_OFF + 400128;                  // 51,600,128 (256-aligned)
static constexpr size_t WC1_OFF = H_OFF + (size_t)NNODES * 128 * 2;  // 77,200,128
static constexpr size_t WC2_OFF = WC1_OFF + 128 * 256 * 2;           // 77,265,664

__global__ void k_zero_f32(float* __restrict__ p, long n) {
    long i = (long)blockIdx.x * blockDim.x + threadIdx.x;
    long s = (long)gridDim.x * blockDim.x;
    for (; i < n; i += s) p[i] = 0.0f;
}

__global__ void k_wcat(const float* __restrict__ Wl, const float* __restrict__ Wr,
                       _Float16* __restrict__ wcat, int nOut) {
    int i = blockIdx.x * blockDim.x + threadIdx.x;
    if (i >= nOut * KTOT) return;
    int n = i >> 8, k = i & 255;
    float v = (k < 128) ? Wl[n * 128 + k] : Wr[n * 128 + (k - 128)];
    wcat[i] = (_Float16)v;
}

__global__ void k_count(const int* __restrict__ tgt, float* __restrict__ cnt) {
    int e = blockIdx.x * blockDim.x + threadIdx.x;
    if (e < NEDGES) atomicAdd(&cnt[tgt[e]], 1.0f);
}

// one wave32 per edge; each lane moves 4 channels (coalesced 16B/8B loads, f32 atomics to L2)
template <bool FEAT_HALF>
__global__ void k_scatter(const void* __restrict__ feat_, const int* __restrict__ src,
                          const int* __restrict__ tgt, float* __restrict__ agg) {
    int warp = blockIdx.x * (blockDim.x >> 5) + (threadIdx.x >> 5);
    int lane = threadIdx.x & 31;
    if (warp >= NEDGES) return;
    int s = src[warp], t = tgt[warp];
    int c0 = lane * 4;
    float v0, v1, v2, v3;
    if (FEAT_HALF) {
        v4h q = *(const v4h*)((const _Float16*)feat_ + (long)s * 128 + c0);
        v0 = (float)q[0]; v1 = (float)q[1]; v2 = (float)q[2]; v3 = (float)q[3];
    } else {
        float4 q = *(const float4*)((const float*)feat_ + (long)s * 128 + c0);
        v0 = q.x; v1 = q.y; v2 = q.z; v3 = q.w;
    }
    float* a = agg + (long)t * 128 + c0;
    atomicAdd(a + 0, v0); atomicAdd(a + 1, v1);
    atomicAdd(a + 2, v2); atomicAdd(a + 3, v3);
}

__device__ __forceinline__ v16h cat8(v8h lo, v8h hi) {
    v16h r;
#pragma unroll
    for (int i = 0; i < 8; ++i) { r[i] = lo[i]; r[8 + i] = hi[i]; }
    return r;
}

// Fused SAGE layer: D[16 x NT*16] = [mean | feat](16x256) @ wcat^T + bias (+ReLU)
// block = (32, NT); wave y computes output cols [16y, 16y+16); one 16-node M-tile per block.
template <int NT, bool FEAT_HALF, bool RELU, typename OutT>
__global__ void k_sage_wmma(const float* __restrict__ agg, const float* __restrict__ cnt,
                            const void* __restrict__ feat_, const _Float16* __restrict__ wcat,
                            const float* __restrict__ bias, OutT* __restrict__ out) {
    constexpr int NOUT = NT * 16;
    __shared__ __align__(16) _Float16 smA[16][A_STRIDE];

    const int  tid      = threadIdx.y * 32 + threadIdx.x;
    const long nodeBase = (long)blockIdx.x * 16;

    // stage A-tile [16 x 256] as f16: cols 0..127 = agg/max(cnt,1), cols 128..255 = feat
    for (int idx = tid; idx < 16 * KTOT; idx += 32 * NT) {
        int  r = idx >> 8, c = idx & 255;
        long node = nodeBase + r;
        float v;
        if (c < 128) {
            float d = cnt[node];
            d = d > 1.0f ? d : 1.0f;
            v = agg[node * 128 + c] / d;
        } else {
            v = FEAT_HALF ? (float)((const _Float16*)feat_)[node * 128 + (c - 128)]
                          : ((const float*)feat_)[node * 128 + (c - 128)];
        }
        smA[r][c] = (_Float16)v;
    }
    __syncthreads();

    const int lane  = threadIdx.x;
    const int nLoc  = lane & 15;            // A row (M) and B col (N) supplied by this lane
    const int kHalf = (lane >> 4) * 8;      // 0 or 8: ISA 16-bit A/B lane split
    const int nGlob = threadIdx.y * 16 + nLoc;
    const _Float16* wrow = wcat + (long)nGlob * KTOT;

    v8f acc = {};
#pragma unroll
    for (int k0 = 0; k0 < KTOT; k0 += 32) {
        v8h a_lo = *(const v8h*)&smA[nLoc][k0 + kHalf];
        v8h a_hi = *(const v8h*)&smA[nLoc][k0 + 16 + kHalf];
        v8h b_lo = *(const v8h*)(wrow + k0 + kHalf);
        v8h b_hi = *(const v8h*)(wrow + k0 + 16 + kHalf);
        v16h a = cat8(a_lo, a_hi);
        v16h b = cat8(b_lo, b_hi);
        acc = __builtin_amdgcn_wmma_f32_16x16x32_f16(
            /*neg_a=*/false, a, /*neg_b=*/false, b,
            /*c_mod=*/(short)0, acc, /*reuse_a=*/false, /*reuse_b=*/false);
    }

    // D layout: VGPR r -> M = r + 8*(lane>>4), N = lane&15
    const float bv    = bias[nGlob];
    const int   mBase = (lane >> 4) * 8;
#pragma unroll
    for (int r = 0; r < 8; ++r) {
        long node = nodeBase + mBase + r;
        float v = acc[r] + bv;
        if (RELU) v = v > 0.0f ? v : 0.0f;
        out[node * NOUT + nGlob] = (OutT)v;
    }
}

extern "C" void kernel_launch(void* const* d_in, const int* in_sizes, int n_in,
                              void* d_out, int out_size, void* d_ws, size_t ws_size,
                              hipStream_t stream) {
    (void)in_sizes; (void)n_in; (void)out_size; (void)ws_size;
    const float* x   = (const float*)d_in[0];
    const int*   ei  = (const int*)d_in[1];
    const float* W1l = (const float*)d_in[2];
    const float* b1  = (const float*)d_in[3];
    const float* W1r = (const float*)d_in[4];
    const float* W2l = (const float*)d_in[5];
    const float* b2  = (const float*)d_in[6];
    const float* W2r = (const float*)d_in[7];
    const int* src = ei;
    const int* tgt = ei + NEDGES;

    char*     ws  = (char*)d_ws;
    float*    agg = (float*)(ws + AGG_OFF);
    float*    cnt = (float*)(ws + CNT_OFF);
    _Float16* h   = (_Float16*)(ws + H_OFF);
    _Float16* wc1 = (_Float16*)(ws + WC1_OFF);
    _Float16* wc2 = (_Float16*)(ws + WC2_OFF);

    // pack weights to f16 [N][256] = [W_l | W_r]
    k_wcat<<<(HIDC * KTOT + 255) / 256, 256, 0, stream>>>(W1l, W1r, wc1, HIDC);
    k_wcat<<<(OUTC * KTOT + 255) / 256, 256, 0, stream>>>(W2l, W2r, wc2, OUTC);

    // ---- layer 1 ----
    k_zero_f32<<<2048, 256, 0, stream>>>(agg, (long)NNODES * 128);
    k_zero_f32<<<512, 256, 0, stream>>>(cnt, (long)NNODES);
    k_count<<<(NEDGES + 255) / 256, 256, 0, stream>>>(tgt, cnt);
    k_scatter<false><<<NEDGES / 8, 256, 0, stream>>>(x, src, tgt, agg);
    k_sage_wmma<8, false, true, _Float16>
        <<<NNODES / 16, dim3(32, 8), 0, stream>>>(agg, cnt, x, wc1, b1, h);

    // ---- layer 2 ----
    k_zero_f32<<<2048, 256, 0, stream>>>(agg, (long)NNODES * 128);
    k_scatter<true><<<NEDGES / 8, 256, 0, stream>>>(h, src, tgt, agg);
    k_sage_wmma<4, true, false, float>
        <<<NNODES / 16, dim3(32, 4), 0, stream>>>(agg, cnt, h, wc2, b2, (float*)d_out);
}
